// TransformerEncoder_24781961298136
// MI455X (gfx1250) — matrix operand order repaired and checked by reading, not compiled, not run on hardware
//
#include <hip/hip_runtime.h>
#include <hip/hip_bf16.h>

// ---------------------------------------------------------------------------
// Transformer encoder layer forward for gfx1250 (wave32).
// bf16 WMMA (v_wmma_f32_16x16x32_bf16) everywhere, activations/weights kept in
// bf16 between GEMMs, async global->LDS staging (ASYNCcnt path), fused
// flash-attention. Conflict-free padded LDS tiles (row stride 144B).
// ---------------------------------------------------------------------------

typedef __attribute__((ext_vector_type(16))) __bf16 v16bf;
typedef __attribute__((ext_vector_type(8)))  float  v8f;
typedef __attribute__((ext_vector_type(4)))  int    i32x4;

union alignas(32) Frag {
    unsigned u[8];
    uint4    q[2];
    v16bf    v;
};

constexpr int Bc   = 2;
constexpr int Tc   = 2048;
constexpr int Dc   = 1024;
constexpr int Hc   = 16;
constexpr int DKc  = 64;
constexpr int DFFc = 4096;

__device__ __forceinline__ unsigned pack_bf16x2(float a, float b) {
    unsigned ua = __builtin_bit_cast(unsigned, a);
    unsigned ub = __builtin_bit_cast(unsigned, b);
    ua += 0x7FFFu + ((ua >> 16) & 1u);   // round to nearest even
    ub += 0x7FFFu + ((ub >> 16) & 1u);
    return (ua >> 16) | (ub & 0xFFFF0000u);
}

// --------------------------- async global->LDS -----------------------------
#if defined(__AMDGCN__) && __has_builtin(__builtin_amdgcn_global_load_async_to_lds_b128)
#define HAVE_ASYNC_LDS 1
typedef __attribute__((address_space(1))) i32x4 as1_i32x4;
typedef __attribute__((address_space(3))) i32x4 as3_i32x4;
#else
#define HAVE_ASYNC_LDS 0
#endif

__device__ __forceinline__ void copy16_g2lds(unsigned* lds_dst, const unsigned* g_src) {
#if HAVE_ASYNC_LDS
    __builtin_amdgcn_global_load_async_to_lds_b128(
        (as1_i32x4*)(g_src), (as3_i32x4*)(lds_dst), 0, 0);
#else
    *(uint4*)lds_dst = *(const uint4*)g_src;
#endif
}

__device__ __forceinline__ void stage_join() {
#if HAVE_ASYNC_LDS
#if __has_builtin(__builtin_amdgcn_s_wait_asynccnt)
    __builtin_amdgcn_s_wait_asynccnt(0);
#else
    asm volatile("s_wait_asynccnt 0x0" ::: "memory");
#endif
#endif
    __syncthreads();
}

// ---------------------------------------------------------------------------
// fp32 -> packed bf16 pair conversion (for weights)
// ---------------------------------------------------------------------------
__global__ __launch_bounds__(256) void cvt_bf16_kernel(
    const float* __restrict__ src, unsigned* __restrict__ dst, int npairs)
{
    const int i = blockIdx.x * 256 + threadIdx.x;
    if (i < npairs) {
        float2 v = *(const float2*)(src + 2 * i);
        dst[i] = pack_bf16x2(v.x, v.y);
    }
}

// ---------------------------------------------------------------------------
// LayerNorm over D=1024; one 256-thread block per row.
// Writes fp32 (for residual use) and optionally packed-bf16 (GEMM input).
// ---------------------------------------------------------------------------
__global__ __launch_bounds__(256) void ln_kernel(
    const float* __restrict__ x, const float* __restrict__ g,
    const float* __restrict__ be, float* __restrict__ out,
    unsigned* __restrict__ ob)
{
    __shared__ float red[16];
    const int row = blockIdx.x;
    const int tid = threadIdx.x;
    const float* xr = x + (size_t)row * Dc;

    float4 xv = *(const float4*)(xr + tid * 4);
    float s  = xv.x + xv.y + xv.z + xv.w;
    float s2 = xv.x * xv.x + xv.y * xv.y + xv.z * xv.z + xv.w * xv.w;
    for (int off = 1; off < 32; off <<= 1) {
        s  += __shfl_xor(s,  off);
        s2 += __shfl_xor(s2, off);
    }
    const int wave = tid >> 5, lane = tid & 31;
    if (lane == 0) { red[wave] = s; red[8 + wave] = s2; }
    __syncthreads();
    if (tid < 32) {
        float a  = (lane < 8) ? red[lane] : 0.0f;
        float b2 = (lane < 8) ? red[8 + lane] : 0.0f;
        for (int off = 1; off < 8; off <<= 1) {
            a  += __shfl_xor(a,  off);
            b2 += __shfl_xor(b2, off);
        }
        if (lane == 0) { red[0] = a; red[8] = b2; }
    }
    __syncthreads();
    const float mean = red[0] * (1.0f / Dc);
    const float var  = red[8] * (1.0f / Dc) - mean * mean;
    const float rstd = rsqrtf(var + 1e-12f);

    const float4 gv = *(const float4*)(g  + tid * 4);
    const float4 bv = *(const float4*)(be + tid * 4);
    float4 o;
    o.x = (xv.x - mean) * rstd * gv.x + bv.x;
    o.y = (xv.y - mean) * rstd * gv.y + bv.y;
    o.z = (xv.z - mean) * rstd * gv.z + bv.z;
    o.w = (xv.w - mean) * rstd * gv.w + bv.w;
    *(float4*)(out + (size_t)row * Dc + tid * 4) = o;
    if (ob) {
        unsigned* orow = ob + (size_t)row * (Dc / 2);
        orow[tid * 2]     = pack_bf16x2(o.x, o.y);
        orow[tid * 2 + 1] = pack_bf16x2(o.z, o.w);
    }
}

// ---------------------------------------------------------------------------
// GEMM: epilogue(A[M,K] @ W[N,K]^T + bias[N]).  A, W are packed-bf16 pairs
// (row stride K/2 uints). 128x256 block tile, 8 waves (2x4), 64x64 per wave,
// K step 64 -> 32 WMMAs per barrier pair, 1:1 ds_load_b128 : wmma.
// mode: 0 f32 out, 1 bf16 out + relu, 2 f32 out += R, 3 bf16 scatter [B,H,T,DK]
// ---------------------------------------------------------------------------
__global__ __launch_bounds__(256) void gemm_bf16_kernel(
    const unsigned* __restrict__ A, const unsigned* __restrict__ W,
    const float* __restrict__ bias, const float* __restrict__ R,
    float* __restrict__ C, unsigned* __restrict__ Cb,
    int M, int N, int K, int mode)
{
    __shared__ unsigned As[128][36];   // padded: 144B row stride, bank-conflict-free
    __shared__ unsigned Bs[256][36];

    const int tid  = threadIdx.x;
    const int lane = tid & 31;
    const int wave = tid >> 5;
    const int ln   = lane & 15;
    const int kh   = lane >> 4;
    const int wrow = (wave >> 2) * 64;   // 0 / 64
    const int wcol = (wave & 3)  * 64;   // 0,64,128,192
    const int m0   = blockIdx.y * 128;
    const int n0   = blockIdx.x * 256;
    const int Khalf = K >> 1;

    v8f acc[4][4] = {};

    for (int k0u = 0; k0u < Khalf; k0u += 32) {   // 64 K elements per step
        __syncthreads();
        for (int i = 0; i < 4; i++) {              // As: 128 rows x 8 uint4
            const int idx = i * 256 + tid;
            const int row = idx >> 3, q4 = idx & 7;
            copy16_g2lds(&As[row][q4 * 4],
                         A + (size_t)(m0 + row) * Khalf + k0u + q4 * 4);
        }
        for (int i = 0; i < 8; i++) {              // Bs: 256 rows x 8 uint4
            const int idx = i * 256 + tid;
            const int row = idx >> 3, q4 = idx & 7;
            copy16_g2lds(&Bs[row][q4 * 4],
                         W + (size_t)(n0 + row) * Khalf + k0u + q4 * 4);
        }
        stage_join();

        for (int c = 0; c < 2; c++) {              // two K=32 chunks
            Frag a[4];
            for (int mi = 0; mi < 4; mi++) {
                const unsigned* ar = &As[wrow + mi * 16 + ln][c * 16];
                a[mi].q[0] = *(const uint4*)(ar + kh * 4);
                a[mi].q[1] = *(const uint4*)(ar + 8 + kh * 4);
            }
            for (int ni = 0; ni < 4; ni++) {
                Frag b;
                const unsigned* br = &Bs[wcol + ni * 16 + ln][c * 16];
                b.q[0] = *(const uint4*)(br + kh * 4);
                b.q[1] = *(const uint4*)(br + 8 + kh * 4);
                for (int mi = 0; mi < 4; mi++)
                    acc[mi][ni] = __builtin_amdgcn_wmma_f32_16x16x32_bf16(
                        false, a[mi].v, false, b.v, (short)0, acc[mi][ni],
                        false, false);
            }
        }
    }

    for (int mi = 0; mi < 4; mi++)
        for (int ni = 0; ni < 4; ni++) {
            const int gn = n0 + wcol + ni * 16 + ln;
            const float bv = bias[gn];
            for (int r = 0; r < 8; r++) {
                const int gm = m0 + wrow + mi * 16 + r + 8 * kh;
                float v = acc[mi][ni][r] + bv;
                if (mode == 2) v += R[(size_t)gm * N + gn];
                if (mode == 0 || mode == 2) {
                    C[(size_t)gm * N + gn] = v;
                } else {
                    if (mode == 1) v = fmaxf(v, 0.0f);
                    const float nb = __shfl_xor(v, 1);
                    if ((ln & 1) == 0) {
                        const unsigned p = pack_bf16x2(v, nb);
                        if (mode == 3) {
                            const int bb = gm >> 11, t = gm & (Tc - 1);
                            const int h  = gn >> 6,  dk = gn & 63;
                            Cb[(((size_t)bb * Hc + h) * Tc + t) * 32 + (dk >> 1)] = p;
                        } else {
                            Cb[((size_t)gm * N + gn) >> 1] = p;
                        }
                    }
                }
            }
        }
}

// ---------------------------------------------------------------------------
// Fused flash attention; Q,K,V packed bf16 in [B,H,T,64]. One block per
// (128-query block, bh). 8 waves x 16 query rows; key blocks of 64.
// Output packed bf16 scattered to [B,T,D].
// ---------------------------------------------------------------------------
__global__ __launch_bounds__(256) void attn_kernel(
    const unsigned* __restrict__ Q, const unsigned* __restrict__ Kg,
    const unsigned* __restrict__ V, const int* __restrict__ mask,
    unsigned* __restrict__ Ob)
{
    __shared__ unsigned Qs[128][36];      // [q row][d/2]  (padded rows)
    __shared__ unsigned Ks[64][36];       // [key row][d/2]
    __shared__ unsigned Vts[64][36];      // [d][key/2] (V transposed)
    __shared__ unsigned Ps[8][16][36];    // per-wave P tile [row][key/2]

    const int tid  = threadIdx.x;
    const int lane = tid & 31;
    const int wave = tid >> 5;
    const int ln   = lane & 15;
    const int kh   = lane >> 4;
    const int qb   = blockIdx.x;          // 0..15
    const int bh   = blockIdx.y;          // 0..31
    const size_t baseU = (size_t)bh * Tc * (DKc / 2);
    const unsigned* Qp = Q + baseU + (size_t)qb * 128 * (DKc / 2);
    const unsigned short* Vh = (const unsigned short*)V;
    const size_t baseE = (size_t)bh * Tc * DKc;

    for (int i = 0; i < 4; i++) {                  // Qs: 128 rows x 8 uint4
        const int idx = i * 256 + tid;
        const int row = idx >> 3, q4 = idx & 7;
        copy16_g2lds(&Qs[row][q4 * 4], Qp + row * 32 + q4 * 4);
    }

    v8f acc[4] = {};
    float mrow[8], lrow[8];
    for (int r = 0; r < 8; r++) { mrow[r] = -3.0e38f; lrow[r] = 0.0f; }

    for (int j = 0; j < Tc / 64; j++) {
        __syncthreads();
        for (int i = 0; i < 2; i++) {              // Ks: 64 rows x 8 uint4
            const int idx = i * 256 + tid;
            const int row = idx >> 3, q4 = idx & 7;
            copy16_g2lds(&Ks[row][q4 * 4],
                         Kg + baseU + (size_t)(j * 64 + row) * 32 + q4 * 4);
        }
        for (int i = 0; i < 8; i++) {              // Vt: 64x64 -> 64 rows x 32 uints
            const int idx = i * 256 + tid;
            const int row = idx >> 5, u = idx & 31;     // row = d, u = key pair
            const unsigned lo = Vh[baseE + (size_t)(j * 64 + 2 * u)     * DKc + row];
            const unsigned hi = Vh[baseE + (size_t)(j * 64 + 2 * u + 1) * DKc + row];
            Vts[row][u] = lo | (hi << 16);
        }
        stage_join();

        // S = Q K^T for this wave's 16 rows x 64 keys
        v8f s[4] = {};
        Frag a, b;
        const unsigned* qr = &Qs[wave * 16 + ln][0];
        for (int c = 0; c < 2; c++) {
            a.q[0] = *(const uint4*)(qr + c * 16 + kh * 4);
            a.q[1] = *(const uint4*)(qr + c * 16 + 8 + kh * 4);
            for (int ni = 0; ni < 4; ni++) {
                const unsigned* kr = &Ks[ni * 16 + ln][0];
                b.q[0] = *(const uint4*)(kr + c * 16 + kh * 4);
                b.q[1] = *(const uint4*)(kr + c * 16 + 8 + kh * 4);
                s[ni] = __builtin_amdgcn_wmma_f32_16x16x32_bf16(
                    false, a.v, false, b.v, (short)0, s[ni], false, false);
            }
        }

        // scale + mask
        const int gq0 = qb * 128 + wave * 16 + 8 * kh;
        for (int ni = 0; ni < 4; ni++) {
            const int gk = j * 64 + ni * 16 + ln;
            for (int r = 0; r < 8; r++) {
                float val = s[ni][r] * 0.125f;
                if (mask[(size_t)(gq0 + r) * Tc + gk] == 0) val = -1.0e9f;
                s[ni][r] = val;
            }
        }

        // online softmax update
        for (int r = 0; r < 8; r++) {
            float mx = s[0][r];
            for (int ni = 1; ni < 4; ni++) mx = fmaxf(mx, s[ni][r]);
            for (int off = 1; off < 16; off <<= 1) mx = fmaxf(mx, __shfl_xor(mx, off));
            const float nm = fmaxf(mrow[r], mx);
            const float alpha = __expf(mrow[r] - nm);
            mrow[r] = nm;
            float rs = 0.0f;
            for (int ni = 0; ni < 4; ni++) {
                const float p = __expf(s[ni][r] - nm);
                s[ni][r] = p;
                rs += p;
            }
            for (int off = 1; off < 16; off <<= 1) rs += __shfl_xor(rs, off);
            lrow[r] = lrow[r] * alpha + rs;
            for (int ni = 0; ni < 4; ni++) acc[ni][r] *= alpha;
        }

        // P -> LDS as packed bf16 pairs (pair partner via lane shuffle)
        for (int ni = 0; ni < 4; ni++)
            for (int r = 0; r < 8; r++) {
                const float me = s[ni][r];
                const float nb = __shfl_xor(me, 1);
                if ((ln & 1) == 0)
                    Ps[wave][r + 8 * kh][(ni * 16 + ln) >> 1] = pack_bf16x2(me, nb);
            }
        __syncthreads();

        // O += P @ V  (A = P 16x64, W = V^T rows)
        const unsigned* pr = &Ps[wave][ln][0];
        for (int c = 0; c < 2; c++) {
            a.q[0] = *(const uint4*)(pr + c * 16 + kh * 4);
            a.q[1] = *(const uint4*)(pr + c * 16 + 8 + kh * 4);
            for (int ni = 0; ni < 4; ni++) {
                const unsigned* vr = &Vts[ni * 16 + ln][0];
                b.q[0] = *(const uint4*)(vr + c * 16 + kh * 4);
                b.q[1] = *(const uint4*)(vr + c * 16 + 8 + kh * 4);
                acc[ni] = __builtin_amdgcn_wmma_f32_16x16x32_bf16(
                    false, a.v, false, b.v, (short)0, acc[ni], false, false);
            }
        }
    }

    // normalize, pack to bf16, scatter: att2d[(b*T + t)*D + h*64 + d]
    const int bI = bh >> 4, hI = bh & 15;
    for (int ni = 0; ni < 4; ni++) {
        const int d = ni * 16 + ln;
        for (int r = 0; r < 8; r++) {
            const int t = qb * 128 + wave * 16 + r + 8 * kh;
            const float v = acc[ni][r] / lrow[r];
            const float nb = __shfl_xor(v, 1);
            if ((ln & 1) == 0)
                Ob[(((size_t)bI * Tc + t) * Dc + hI * 64 + d) >> 1] =
                    pack_bf16x2(v, nb);
        }
    }
}

// ---------------------------------------------------------------------------
extern "C" void kernel_launch(void* const* d_in, const int* in_sizes, int n_in,
                              void* d_out, int out_size, void* d_ws, size_t ws_size,
                              hipStream_t stream) {
    const float* x     = (const float*)d_in[0];
    const int*   mask  = (const int*)  d_in[1];
    const float* wq    = (const float*)d_in[2];
    const float* bq    = (const float*)d_in[3];
    const float* wk    = (const float*)d_in[4];
    const float* bk    = (const float*)d_in[5];
    const float* wv    = (const float*)d_in[6];
    const float* bv    = (const float*)d_in[7];
    const float* wo    = (const float*)d_in[8];
    const float* bo    = (const float*)d_in[9];
    const float* w1    = (const float*)d_in[10];
    const float* b1    = (const float*)d_in[11];
    const float* w2    = (const float*)d_in[12];
    const float* b2    = (const float*)d_in[13];
    const float* g1    = (const float*)d_in[14];
    const float* beta1 = (const float*)d_in[15];
    const float* g2    = (const float*)d_in[16];
    const float* beta2 = (const float*)d_in[17];
    float* out = (float*)d_out;
    float* ws  = (float*)d_ws;

    const int M = Bc * Tc;                         // 4096
    const size_t MF = (size_t)M * Dc;              // 4M floats

    // fp32 region
    float* xn  = ws;                               // residual for o-proj
    float* x2  = ws + 1 * MF;
    float* xln = ws + 2 * MF;                      // residual for final add
    // packed-bf16 region (uints)
    unsigned* ub   = (unsigned*)(ws + 3 * MF);
    const size_t CU = MF / 2;                      // 2M uints per [M,D] bf16
    unsigned* xnb  = ub;
    unsigned* xlnb = ub + 1 * CU;
    unsigned* qb_  = ub + 2 * CU;
    unsigned* kb_  = ub + 3 * CU;
    unsigned* vb_  = ub + 4 * CU;
    unsigned* attb = ub + 5 * CU;
    unsigned* ffb  = ub + 6 * CU;                  // M x DFF -> 8M uints
    unsigned* wqb  = ub + 10 * CU;
    unsigned* wkb  = wqb + (size_t)Dc * Dc / 2;
    unsigned* wvb  = wkb + (size_t)Dc * Dc / 2;
    unsigned* wob  = wvb + (size_t)Dc * Dc / 2;
    unsigned* w1b  = wob + (size_t)Dc * Dc / 2;
    unsigned* w2b  = w1b + (size_t)DFFc * Dc / 2;

    const dim3 blk(256);
    const int npD  = Dc * Dc / 2;                  // 512K pairs
    const int npFF = DFFc * Dc / 2;                // 2M pairs

    cvt_bf16_kernel<<<npD  / 256, blk, 0, stream>>>(wq, wqb, npD);
    cvt_bf16_kernel<<<npD  / 256, blk, 0, stream>>>(wk, wkb, npD);
    cvt_bf16_kernel<<<npD  / 256, blk, 0, stream>>>(wv, wvb, npD);
    cvt_bf16_kernel<<<npD  / 256, blk, 0, stream>>>(wo, wob, npD);
    cvt_bf16_kernel<<<npFF / 256, blk, 0, stream>>>(w1, w1b, npFF);
    cvt_bf16_kernel<<<npFF / 256, blk, 0, stream>>>(w2, w2b, npFF);

    ln_kernel<<<M, blk, 0, stream>>>(x, g1, beta1, xn, xnb);

    const dim3 gD(Dc / 256, M / 128);              // (4, 32)
    const dim3 gF(DFFc / 256, M / 128);            // (16, 32)

    gemm_bf16_kernel<<<gD, blk, 0, stream>>>(xnb, wqb, bq, nullptr, nullptr, qb_, M, Dc, Dc, 3);
    gemm_bf16_kernel<<<gD, blk, 0, stream>>>(xnb, wkb, bk, nullptr, nullptr, kb_, M, Dc, Dc, 3);
    gemm_bf16_kernel<<<gD, blk, 0, stream>>>(xnb, wvb, bv, nullptr, nullptr, vb_, M, Dc, Dc, 3);

    attn_kernel<<<dim3(Tc / 128, Bc * Hc), blk, 0, stream>>>(qb_, kb_, vb_, mask, attb);

    // x2 = xn + (att @ wo^T + bo)
    gemm_bf16_kernel<<<gD, blk, 0, stream>>>(attb, wob, bo, xn, x2, nullptr, M, Dc, Dc, 2);

    ln_kernel<<<M, blk, 0, stream>>>(x2, g2, beta2, xln, xlnb);

    // ff = relu(xln @ w1^T + b1), bf16
    gemm_bf16_kernel<<<gF, blk, 0, stream>>>(xlnb, w1b, b1, nullptr, nullptr, ffb, M, DFFc, Dc, 1);
    // out = xln + (ff @ w2^T + b2)
    gemm_bf16_kernel<<<gD, blk, 0, stream>>>(ffb, w2b, b2, xln, out, nullptr, M, Dc, DFFc, 2);
}
